// AttendAndSpeller_34617436405770
// MI455X (gfx1250) — matrix operand order repaired and checked by reading, not compiled, not run on hardware
//
#include <hip/hip_runtime.h>
#include <math.h>

#define Bb 64
#define Ss 1024
#define Hh 1024
#define Vv 32000
#define NCHUNK 8
#define SCHUNK 128

typedef __attribute__((ext_vector_type(16))) __bf16 v16bf;
typedef __attribute__((ext_vector_type(8)))  float  v8f;

// ---------------- CDNA5 WMMA bf16 fragment layout helpers ------------------
// A (16x32, 16-bit): lane 0-15 -> M=lane, lanes 16-31 -> M=lane-16.
// VGPR0-3 hold K = half*8 + 0..7 ; VGPR4-7 hold K = 16 + half*8 + 0..7.
// B (32x16, 16-bit): lane&15 -> N; elements hold K in the same order as A.

__device__ __forceinline__ v16bf cvt4_to_bf16(float4 r0, float4 r1, float4 r2, float4 r3) {
    v16bf v;
    v[0]=(__bf16)r0.x; v[1]=(__bf16)r0.y; v[2]=(__bf16)r0.z; v[3]=(__bf16)r0.w;
    v[4]=(__bf16)r1.x; v[5]=(__bf16)r1.y; v[6]=(__bf16)r1.z; v[7]=(__bf16)r1.w;
    v[8]=(__bf16)r2.x; v[9]=(__bf16)r2.y; v[10]=(__bf16)r2.z; v[11]=(__bf16)r2.w;
    v[12]=(__bf16)r3.x; v[13]=(__bf16)r3.y; v[14]=(__bf16)r3.z; v[15]=(__bf16)r3.w;
    return v;
}

// Hot GEMM: D[MxN] = A@B (+ A2@B2) + bias; each wave owns a 16x64 strip.
// Per k-iteration: issue all 4 b128 A loads + 64 dword B loads into distinct locals (one clause,
// staged loadcnt waits), then convert, then 4 WMMAs sharing one A fragment.
// __launch_bounds__(256,1): blocks are 8 waves -> allocator may use the full
// per-wave VGPR budget (~170 live regs here) without spilling accumulators.
__global__ void __launch_bounds__(256, 1)
wmma_gemm4(const float* __restrict__ A, int lda,
           const float* __restrict__ B, int ldb, int K,
           const float* __restrict__ A2, int lda2,
           const float* __restrict__ B2, int ldb2, int K2,
           const float* __restrict__ bias,
           float* __restrict__ D, int ldd, int M, int N) {
    int lane = threadIdx.x & 31;
    int wave = threadIdx.x >> 5;
    int nquads = N >> 6;                       // strips of 64 columns
    int total = (M >> 4) * nquads;
    int task = blockIdx.x * (blockDim.x >> 5) + wave;
    if (task >= total) return;                 // wave-uniform exit (EXEC stays all-1)
    int m0 = (task / nquads) << 4;
    int n0 = (task % nquads) << 6;

    v8f acc[4];
    #pragma unroll
    for (int nt = 0; nt < 4; ++nt)
        acc[nt] = (v8f){0.f, 0.f, 0.f, 0.f, 0.f, 0.f, 0.f, 0.f};

    int half8 = (lane >> 4) << 3;              // A: 0 or 8
    const float* arow = A + (size_t)(m0 + (lane & 15)) * lda + half8;

    for (int k0 = 0; k0 < K; k0 += 32) {
        // ---- raw loads: distinct destinations so they clause together ----
        const float* ap = arow + k0;
        float4 a0 = *(const float4*)(ap);
        float4 a1 = *(const float4*)(ap + 4);
        float4 a2 = *(const float4*)(ap + 16);
        float4 a3 = *(const float4*)(ap + 20);
        const float* bp = B + (size_t)(k0 + half8) * ldb + n0 + (lane & 15);
        float4 br[16];
        #pragma unroll
        for (int j = 0; j < 16; ++j) {
            const float* p = bp + (size_t)(((j & 1) << 2) + (((j >> 1) & 1) << 4)) * ldb + (j >> 2) * 16;
            br[j] = make_float4(p[0], p[(size_t)ldb], p[(size_t)2 * ldb], p[(size_t)3 * ldb]);
        }
        if (k0 + 32 < K)                       // keep the weight stream ahead
            __builtin_prefetch((const void*)(bp + (size_t)32 * ldb), 0, 0);
        // ---- convert ----
        v16bf a = cvt4_to_bf16(a0, a1, a2, a3);
        v16bf bfr[4];
        #pragma unroll
        for (int nt = 0; nt < 4; ++nt)
            bfr[nt] = cvt4_to_bf16(br[nt*4+0], br[nt*4+1], br[nt*4+2], br[nt*4+3]);
        // ---- 4 WMMAs reuse one A fragment ----
        #pragma unroll
        for (int nt = 0; nt < 4; ++nt)
            acc[nt] = __builtin_amdgcn_wmma_f32_16x16x32_bf16(false, a, false, bfr[nt],
                                                              (short)0, acc[nt], false, false);
    }
    if (A2) {
        const float* arow2 = A2 + (size_t)(m0 + (lane & 15)) * lda2 + half8;
        for (int k0 = 0; k0 < K2; k0 += 32) {
            const float* ap = arow2 + k0;
            float4 a0 = *(const float4*)(ap);
            float4 a1 = *(const float4*)(ap + 4);
            float4 a2 = *(const float4*)(ap + 16);
            float4 a3 = *(const float4*)(ap + 20);
            const float* bp = B2 + (size_t)(k0 + half8) * ldb2 + n0 + (lane & 15);
            float4 br[16];
            #pragma unroll
            for (int j = 0; j < 16; ++j) {
                const float* p = bp + (size_t)(((j & 1) << 2) + (((j >> 1) & 1) << 4)) * ldb2 + (j >> 2) * 16;
                br[j] = make_float4(p[0], p[(size_t)ldb2], p[(size_t)2 * ldb2], p[(size_t)3 * ldb2]);
            }
            if (k0 + 32 < K2)
                __builtin_prefetch((const void*)(bp + (size_t)32 * ldb2), 0, 0);
            v16bf a = cvt4_to_bf16(a0, a1, a2, a3);
            v16bf bfr[4];
            #pragma unroll
            for (int nt = 0; nt < 4; ++nt)
                bfr[nt] = cvt4_to_bf16(br[nt*4+0], br[nt*4+1], br[nt*4+2], br[nt*4+3]);
            #pragma unroll
            for (int nt = 0; nt < 4; ++nt)
                acc[nt] = __builtin_amdgcn_wmma_f32_16x16x32_bf16(false, a, false, bfr[nt],
                                                                  (short)0, acc[nt], false, false);
        }
    }
    // C/D layout: VGPR r -> (M = mbase + r, N = ntile + lane&15)
    int mb = m0 + ((lane >> 4) << 3);
    #pragma unroll
    for (int nt = 0; nt < 4; ++nt) {
        int n = n0 + nt * 16 + (lane & 15);
        float bv = bias ? bias[n] : 0.f;
        #pragma unroll
        for (int r = 0; r < 8; ++r)
            D[(size_t)(mb + r) * ldd + n] = acc[nt][r] + bv;
    }
}

// Small transposed-B GEMM (qk = q @ Wk^T only): one 16x16 tile per wave.
__global__ void __launch_bounds__(256, 1)
wmma_gemm_tn(const float* __restrict__ A, int lda,
             const float* __restrict__ Bt, int ldb, int K,
             float* __restrict__ D, int ldd, int M, int N) {
    int lane = threadIdx.x & 31;
    int wave = threadIdx.x >> 5;
    int ntiles = N >> 4;
    int total = (M >> 4) * ntiles;
    int tile = blockIdx.x * (blockDim.x >> 5) + wave;
    if (tile >= total) return;
    int m0 = (tile / ntiles) << 4;
    int n0 = (tile % ntiles) << 4;

    int half8 = (lane >> 4) << 3;
    const float* arow = A + (size_t)(m0 + (lane & 15)) * lda + half8;

    v8f acc = {0.f, 0.f, 0.f, 0.f, 0.f, 0.f, 0.f, 0.f};
    for (int k0 = 0; k0 < K; k0 += 32) {
        const float* ap = arow + k0;
        float4 a0 = *(const float4*)(ap);
        float4 a1 = *(const float4*)(ap + 4);
        float4 a2 = *(const float4*)(ap + 16);
        float4 a3 = *(const float4*)(ap + 20);
        // logical B[k][n] = Bt[n][k] (Bt row-major, leading dim ldb)
        const float* p = Bt + (size_t)(n0 + (lane & 15)) * ldb + (k0 + half8);
        float raw[16];
        #pragma unroll
        for (int e = 0; e < 16; ++e) raw[e] = p[(e & 7) + ((e >> 3) << 4)];
        v16bf a = cvt4_to_bf16(a0, a1, a2, a3);
        v16bf b;
        #pragma unroll
        for (int e = 0; e < 16; ++e) b[e] = (__bf16)raw[e];
        acc = __builtin_amdgcn_wmma_f32_16x16x32_bf16(false, a, false, b,
                                                      (short)0, acc, false, false);
    }
    int n = n0 + (lane & 15);
    int mb = m0 + ((lane >> 4) << 3);
    #pragma unroll
    for (int r = 0; r < 8; ++r)
        D[(size_t)(mb + r) * ldd + n] = acc[r];
}

// ---------------- qb[b] = q[b,:] . bk -------------------------------------
__global__ void dot_q_bk(const float* __restrict__ q, const float* __restrict__ bk,
                         float* __restrict__ qb) {
    int b = blockIdx.x, lane = threadIdx.x;
    float acc = 0.f;
    for (int j = lane; j < Hh; j += 32) acc += q[b * Hh + j] * bk[j];
    for (int off = 16; off; off >>= 1) acc += __shfl_xor(acc, off, 32);
    if (lane == 0) qb[b] = acc;
}

// ---------------- flash-style attention partials ---------------------------
// grid (NCHUNK, B); each block: scores for a 128-row chunk via qk-dot
// (audio streamed from HBM once), local softmax stats, weighted partial
// context (chunk re-read immediately -> L2 hit).
__global__ void attn_partial(const float* __restrict__ audio,
                             const unsigned char* __restrict__ amask,
                             const float* __restrict__ qk,
                             const float* __restrict__ qb,
                             float* __restrict__ pm, float* __restrict__ pl,
                             float* __restrict__ pctx) {
    __shared__ float qksh[Hh];
    __shared__ float sc[SCHUNK];
    __shared__ float red[8];
    __shared__ float s_max, s_sum;
    int b = blockIdx.y, ch = blockIdx.x;
    int t = threadIdx.x, lane = t & 31, wave = t >> 5;
    int s0 = ch * SCHUNK;
    const float* Ab = audio + (size_t)b * Ss * Hh;

    for (int j = t; j < Hh; j += 256) qksh[j] = qk[b * Hh + j];
    __syncthreads();
    float qbb = qb[b];

    // scores: each wave owns 16 rows, wave-local reduction only
    for (int i = 0; i < SCHUNK / 8; ++i) {
        int sl = wave * (SCHUNK / 8) + i;
        const float* row = Ab + (size_t)(s0 + sl) * Hh;
        float acc = 0.f;
        #pragma unroll 8
        for (int j = lane; j < Hh; j += 32) acc += qksh[j] * row[j];
        for (int off = 16; off; off >>= 1) acc += __shfl_xor(acc, off, 32);
        if (lane == 0) {
            float mm = amask[(size_t)b * Ss + s0 + sl] ? 0.f : 1.f;
            sc[sl] = acc + qbb - 1.0e9f * mm;
        }
    }
    __syncthreads();

    // chunk max
    float lm = -3.4e38f;
    for (int i = t; i < SCHUNK; i += 256) lm = fmaxf(lm, sc[i]);
    for (int off = 16; off; off >>= 1) lm = fmaxf(lm, __shfl_xor(lm, off, 32));
    if (lane == 0) red[wave] = lm;
    __syncthreads();
    if (t == 0) {
        float m = red[0];
        for (int w = 1; w < 8; ++w) m = fmaxf(m, red[w]);
        s_max = m;
    }
    __syncthreads();
    float m = s_max;

    // exp + chunk sum (sc becomes unnormalized probs)
    float ls = 0.f;
    for (int i = t; i < SCHUNK; i += 256) { float p = __expf(sc[i] - m); sc[i] = p; ls += p; }
    for (int off = 16; off; off >>= 1) ls += __shfl_xor(ls, off, 32);
    if (lane == 0) red[wave] = ls;
    __syncthreads();
    if (t == 0) {
        float ss = 0.f;
        for (int w = 0; w < 8; ++w) ss += red[w];
        s_sum = ss;
        pm[b * NCHUNK + ch] = m;
        pl[b * NCHUNK + ch] = ss;
    }
    __syncthreads();
    (void)s_sum;

    // partial context: coalesced along H
    for (int h = t; h < Hh; h += 256) {
        float a = 0.f;
        #pragma unroll 4
        for (int i = 0; i < SCHUNK; ++i) a += sc[i] * Ab[(size_t)(s0 + i) * Hh + h];
        pctx[(size_t)(b * NCHUNK + ch) * Hh + h] = a;
    }
}

// ---------------- combine partials + build xcat = [emb[dec] | context] -----
__global__ void attn_finalize(const float* __restrict__ pm, const float* __restrict__ pl,
                              const float* __restrict__ pctx,
                              const float* __restrict__ emb, const int* __restrict__ dec,
                              float* __restrict__ xcat) {
    __shared__ float wgt[NCHUNK];
    __shared__ float invL;
    int b = blockIdx.x, t = threadIdx.x;
    if (t == 0) {
        float M = -3.4e38f;
        for (int c = 0; c < NCHUNK; ++c) M = fmaxf(M, pm[b * NCHUNK + c]);
        float L = 0.f;
        for (int c = 0; c < NCHUNK; ++c) {
            float e = __expf(pm[b * NCHUNK + c] - M);
            wgt[c] = e;
            L += e * pl[b * NCHUNK + c];
        }
        invL = 1.f / L;
    }
    __syncthreads();
    size_t erow = (size_t)dec[b] * Hh;
    for (int h = t; h < Hh; h += 256) {
        float a = 0.f;
        #pragma unroll
        for (int c = 0; c < NCHUNK; ++c) a += wgt[c] * pctx[(size_t)(b * NCHUNK + c) * Hh + h];
        xcat[(size_t)b * 2 * Hh + Hh + h] = a * invL;
        xcat[(size_t)b * 2 * Hh + h] = emb[erow + h];
    }
}

// ---------------- LSTM gate epilogue (Keras i,f,g,o order + masking) -------
__device__ __forceinline__ float sigf(float x) { return 1.f / (1.f + __expf(-x)); }

__global__ void lstm_gates(const float* __restrict__ z,
                           const float* __restrict__ hprev, const float* __restrict__ cprev,
                           const int* __restrict__ dec,
                           float* __restrict__ h, float* __restrict__ c,
                           float* __restrict__ out) {
    int idx = blockIdx.x * blockDim.x + threadIdx.x;
    if (idx >= Bb * Hh) return;
    int b = idx >> 10;           // / Hh
    int n = idx & (Hh - 1);      // % Hh
    const float* zb = z + (size_t)b * 4 * Hh;
    float gi = sigf(zb[n]);
    float gf = sigf(zb[Hh + n]);
    float gg = tanhf(zb[2 * Hh + n]);
    float go = sigf(zb[3 * Hh + n]);
    float cp = cprev[idx], hp = hprev[idx];
    float cn = gf * cp + gi * gg;
    float hn = go * tanhf(cn);
    bool msk = dec[b] != 0;
    out[idx] = msk ? hn : 0.f;
    h[idx]   = msk ? hn : hp;
    c[idx]   = msk ? cn : cp;
}

// ---------------------------------------------------------------------------
extern "C" void kernel_launch(void* const* d_in, const int* in_sizes, int n_in,
                              void* d_out, int out_size, void* d_ws, size_t ws_size,
                              hipStream_t stream) {
    const float* audio = (const float*)d_in[0];
    const int*   dec   = (const int*)d_in[1];
    const unsigned char* amask = (const unsigned char*)d_in[2];
    const float* h0 = (const float*)d_in[3];
    const float* c0 = (const float*)d_in[4];
    const float* emb = (const float*)d_in[5];
    const float* Wq = (const float*)d_in[6];
    const float* bq = (const float*)d_in[7];
    const float* Wk = (const float*)d_in[8];
    const float* bk = (const float*)d_in[9];
    const float* W0 = (const float*)d_in[10];
    const float* U0 = (const float*)d_in[11];
    const float* b0 = (const float*)d_in[12];
    const float* W1 = (const float*)d_in[13];
    const float* U1 = (const float*)d_in[14];
    const float* b1 = (const float*)d_in[15];
    const float* W2 = (const float*)d_in[16];
    const float* U2 = (const float*)d_in[17];
    const float* b2 = (const float*)d_in[18];
    const float* Wf = (const float*)d_in[19];
    const float* bf = (const float*)d_in[20];

    float* outp   = (float*)d_out;
    float* logits = outp;
    float* h_out  = outp + (size_t)Bb * Vv;
    float* c_out  = h_out + (size_t)Bb * Hh;

    float* ws   = (float*)d_ws;
    float* q    = ws;                 // 65536
    float* qk   = ws + 65536;         // 65536
    float* qb   = ws + 131072;        // 64
    float* pm   = ws + 131136;        // 512
    float* pl   = ws + 131648;        // 512
    float* pctx = ws + 132160;        // 524288
    float* xcat = ws + 656448;        // 131072
    float* z    = ws + 787520;        // 262144
    float* hb0  = ws + 1049664;       // 65536
    float* cb0  = ws + 1115200;       // 65536
    float* ob0  = ws + 1180736;       // 65536
    float* hb1  = ws + 1246272;       // 65536
    float* cb1  = ws + 1311808;       // 65536
    float* ob1  = ws + 1377344;       // 65536
    float* ob2  = ws + 1442880;       // 65536

    // q = h0 @ Wq + bq     (4 m-tiles x 16 n-strips = 64 wave-tasks -> 8 blocks)
    wmma_gemm4<<<8, 256, 0, stream>>>(h0, Hh, Wq, Hh, Hh,
                                      nullptr, 0, nullptr, 0, 0,
                                      bq, q, Hh, Bb, Hh);
    // qk = q @ Wk^T   (score refactor: scores = qk.audio + q.bk)
    wmma_gemm_tn<<<32, 256, 0, stream>>>(q, Hh, Wk, Hh, Hh, qk, Hh, Bb, Hh);
    dot_q_bk<<<Bb, 32, 0, stream>>>(q, bk, qb);

    attn_partial<<<dim3(NCHUNK, Bb), 256, 0, stream>>>(audio, amask, qk, qb, pm, pl, pctx);
    attn_finalize<<<Bb, 256, 0, stream>>>(pm, pl, pctx, emb, dec, xcat);

    // layer 0: z = xcat @ W0 + h0 @ U0 + b0   (4 x 64 strips = 256 tasks -> 32 blocks)
    wmma_gemm4<<<32, 256, 0, stream>>>(xcat, 2 * Hh, W0, 4 * Hh, 2 * Hh,
                                       h0, Hh, U0, 4 * Hh, Hh,
                                       b0, z, 4 * Hh, Bb, 4 * Hh);
    lstm_gates<<<(Bb * Hh) / 256, 256, 0, stream>>>(z, h0, c0, dec, hb0, cb0, ob0);

    // layer 1
    wmma_gemm4<<<32, 256, 0, stream>>>(ob0, Hh, W1, 4 * Hh, Hh,
                                       hb0, Hh, U1, 4 * Hh, Hh,
                                       b1, z, 4 * Hh, Bb, 4 * Hh);
    lstm_gates<<<(Bb * Hh) / 256, 256, 0, stream>>>(z, hb0, cb0, dec, hb1, cb1, ob1);

    // layer 2 (h,c straight into d_out)
    wmma_gemm4<<<32, 256, 0, stream>>>(ob1, Hh, W2, 4 * Hh, Hh,
                                       hb1, Hh, U2, 4 * Hh, Hh,
                                       b2, z, 4 * Hh, Bb, 4 * Hh);
    lstm_gates<<<(Bb * Hh) / 256, 256, 0, stream>>>(z, hb1, cb1, dec, h_out, c_out, ob2);

    // logits = ob2 @ Wf + bf   (4 x 500 strips = 2000 tasks -> 250 blocks)
    wmma_gemm4<<<250, 256, 0, stream>>>(ob2, Hh, Wf, Vv, Hh,
                                        nullptr, 0, nullptr, 0, 0,
                                        bf, logits, Vv, Bb, Vv);
}
